// IncrementalDecoderLM_89275190215317
// MI455X (gfx1250) — matrix operand order repaired and checked by reading, not compiled, not run on hardware
//
#include <hip/hip_runtime.h>
#include <math.h>

// ---- model constants (from reference) ----
#define SMAXC  1024
#define DMODEL 1024
#define NHEAD  16
#define HDIM   64
#define NLAYER 2
#define DFFC   4096
#define WINC   256
#define BATCH  2
#define VOCAB  32000
#define MTOT   (BATCH * SMAXC)   // 2048 rows

typedef __attribute__((ext_vector_type(16))) __bf16 v16bf;
typedef __attribute__((ext_vector_type(8)))  float  v8f;

// ---------------- embedding + positional ----------------
__global__ void k_embed(const int* __restrict__ ids, const float* __restrict__ emb,
                        const float* __restrict__ pos, float* __restrict__ x) {
    int row = blockIdx.x;            // 0..2047  (b*S + s)
    int s   = row % SMAXC;
    int id  = ids[row];
    int d   = threadIdx.x * 4;
    const float4 e = *(const float4*)&emb[(size_t)id * DMODEL + d];
    const float4 p = *(const float4*)&pos[(size_t)s  * DMODEL + d];
    float4 o; o.x = e.x + p.x; o.y = e.y + p.y; o.z = e.z + p.z; o.w = e.w + p.w;
    *(float4*)&x[(size_t)row * DMODEL + d] = o;
}

// ---------------- RMSNorm (one block per row) ----------------
__global__ void k_rms(const float* __restrict__ x, const float* __restrict__ w,
                      float* __restrict__ out) {
    __shared__ float red[256];
    int row = blockIdx.x, t = threadIdx.x;
    const float4 v = *(const float4*)&x[(size_t)row * DMODEL + t * 4];
    red[t] = v.x * v.x + v.y * v.y + v.z * v.z + v.w * v.w;
    __syncthreads();
    for (int s = 128; s > 0; s >>= 1) { if (t < s) red[t] += red[t + s]; __syncthreads(); }
    float inv = rsqrtf(red[0] / (float)DMODEL + 1e-6f);
    const float4 wv = *(const float4*)&w[t * 4];
    float4 o; o.x = v.x * inv * wv.x; o.y = v.y * inv * wv.y;
    o.z = v.z * inv * wv.z; o.w = v.w * inv * wv.w;
    *(float4*)&out[(size_t)row * DMODEL + t * 4] = o;
}

// ---------------- sliding-window attention (one block per (b,h,i)) ----------------
__global__ void k_attn(const float* __restrict__ q, const float* __restrict__ k,
                       const float* __restrict__ v, float* __restrict__ o) {
    __shared__ float qs[HDIM];
    __shared__ float p[256];
    __shared__ float red[256];
    int blk = blockIdx.x;
    int i = blk % SMAXC;
    int h = (blk / SMAXC) % NHEAD;
    int b = blk / (SMAXC * NHEAD);
    int t = threadIdx.x;
    size_t rowbase = ((size_t)(b * SMAXC + i)) * DMODEL + h * HDIM;
    if (t < HDIM) qs[t] = q[rowbase + t];
    __syncthreads();
    int jstart = i - (WINC - 1); if (jstart < 0) jstart = 0;
    int count = i - jstart + 1;  // 1..256
    float s = -1e30f;
    if (t < count) {
        const float* kr = &k[((size_t)(b * SMAXC + jstart + t)) * DMODEL + h * HDIM];
        float acc = 0.f;
        #pragma unroll 8
        for (int d = 0; d < HDIM; ++d) acc += qs[d] * kr[d];
        s = acc * 0.125f;            // 1/sqrt(64)
    }
    red[t] = s; __syncthreads();
    for (int st = 128; st > 0; st >>= 1) { if (t < st) red[t] = fmaxf(red[t], red[t + st]); __syncthreads(); }
    float mx = red[0]; __syncthreads();
    float e = (t < count) ? __expf(s - mx) : 0.f;
    p[t] = e; red[t] = e; __syncthreads();
    for (int st = 128; st > 0; st >>= 1) { if (t < st) red[t] += red[t + st]; __syncthreads(); }
    float inv = 1.f / red[0];
    __syncthreads();
    if (t < HDIM) {
        float acc = 0.f;
        for (int jj = 0; jj < count; ++jj)
            acc += p[jj] * v[((size_t)(b * SMAXC + jstart + jj)) * DMODEL + h * HDIM + t];
        o[rowbase + t] = acc * inv;
    }
}

// ---------------- WMMA bf16 GEMM ----------------
// C[M,N] = act(A[M,K] @ W + bias) (+ res). WLAY=0: W is [K,N]; WLAY=1: W is [N,K].
// Block tile 128x128, 256 threads = 8 waves (4 along M x 2 along N);
// each wave owns 32x64 = 2x4 WMMA accumulators. Register-prefetch pipeline:
// while step k computes its 8 WMMAs, global loads for step k+1 are in flight.

union FragBF { v16bf v; struct { uint4 lo; uint4 hi; } q; };

// ISA 16-bit A layout (16x32): lane m=lane%16, half=lane/16.
// V0-3 = K[half*8..+7], V4-7 = K[16+half*8..+7].
__device__ inline v16bf load_fragA(const __bf16* rowp, int hf) {
    const uint4* p = (const uint4*)rowp;      // row = 32 bf16 = 4 uint4
    FragBF f; f.q.lo = p[hf]; f.q.hi = p[2 + hf]; return f.v;
}
// ISA 16-bit B layout (32x16): lane col n=lane%16, half=lane/16;
// V0-3 = K[half*8..+7], V4-7 = K[16+half*8..+7] (same as A).
__device__ inline v16bf load_fragB(const __bf16* rowp, int hf) {
    const uint4* p = (const uint4*)rowp;
    FragBF f; f.q.lo = p[hf]; f.q.hi = p[2 + hf]; return f.v;
}

template<int WLAY, int ACT, bool RES>
__global__ __launch_bounds__(256) void k_gemm(const float* __restrict__ A,
                                              const float* __restrict__ Wt,
                                              const float* __restrict__ bias,
                                              const float* __restrict__ res,
                                              float* __restrict__ C,
                                              int M, int N, int K) {
    __shared__ __align__(16) __bf16 lA[128 * 32];   // [m][k]  8 KB
    __shared__ __align__(16) __bf16 lB[128 * 32];   // [n][k]  8 KB
    const int t    = threadIdx.x;
    const int bm   = blockIdx.y * 128;
    const int bn   = blockIdx.x * 128;
    const int lane = t & 31;
    const int wave = t >> 5;
    const int wm   = wave & 3;      // 4 waves along M (4*32 = 128)
    const int wn   = wave >> 2;     // 2 waves along N (2*64 = 128)
    const int hrow = lane & 15;
    const int hf   = lane >> 4;

    v8f acc[2][4] = {};
    float4 rA[4], rB[4];

    // staging coordinates (constant per thread)
    const int colA  = (t & 7) * 4;     // A: 8 threads/row, 4 passes of 32 rows
    const int rbase = t >> 3;
    const int nnB0  = (t & 31) * 4;    // WLAY=0: along N, 4 passes of 8 k-rows
    const int kbB0  = t >> 5;
    const int kkB1  = (t & 7) * 4;     // WLAY=1: along K, 4 passes of 32 n-rows
    const int nbB1  = t >> 3;

    auto prefetch = [&](int k0) {
        #pragma unroll
        for (int p = 0; p < 4; ++p)
            rA[p] = *(const float4*)&A[(size_t)(bm + p * 32 + rbase) * K + k0 + colA];
        if (WLAY == 0) {
            #pragma unroll
            for (int p = 0; p < 4; ++p)
                rB[p] = *(const float4*)&Wt[(size_t)(k0 + p * 8 + kbB0) * N + bn + nnB0];
        } else {
            #pragma unroll
            for (int p = 0; p < 4; ++p)
                rB[p] = *(const float4*)&Wt[(size_t)(bn + p * 32 + nbB1) * K + k0 + kkB1];
        }
    };
    auto stage = [&]() {
        #pragma unroll
        for (int p = 0; p < 4; ++p) {
            union { __bf16 b[4]; uint2 u; } pk;
            pk.b[0] = (__bf16)rA[p].x; pk.b[1] = (__bf16)rA[p].y;
            pk.b[2] = (__bf16)rA[p].z; pk.b[3] = (__bf16)rA[p].w;
            *(uint2*)&lA[(p * 32 + rbase) * 32 + colA] = pk.u;
        }
        if (WLAY == 0) {               // transpose fp32 [k][n] -> bf16 LDS [n][k]
            #pragma unroll
            for (int p = 0; p < 4; ++p) {
                int kk = p * 8 + kbB0;
                lB[(nnB0 + 0) * 32 + kk] = (__bf16)rB[p].x;
                lB[(nnB0 + 1) * 32 + kk] = (__bf16)rB[p].y;
                lB[(nnB0 + 2) * 32 + kk] = (__bf16)rB[p].z;
                lB[(nnB0 + 3) * 32 + kk] = (__bf16)rB[p].w;
            }
        } else {                       // already [n][k]: packed copy
            #pragma unroll
            for (int p = 0; p < 4; ++p) {
                union { __bf16 b[4]; uint2 u; } pk;
                pk.b[0] = (__bf16)rB[p].x; pk.b[1] = (__bf16)rB[p].y;
                pk.b[2] = (__bf16)rB[p].z; pk.b[3] = (__bf16)rB[p].w;
                *(uint2*)&lB[(p * 32 + nbB1) * 32 + kkB1] = pk.u;
            }
        }
    };

    prefetch(0);
    for (int k0 = 0; k0 < K; k0 += 32) {
        __syncthreads();               // all reads of LDS from previous step done
        stage();
        __syncthreads();               // tiles visible
        if (k0 + 32 < K) prefetch(k0 + 32);   // overlaps with WMMAs below

        v16bf af[2], bf[4];
        #pragma unroll
        for (int tm = 0; tm < 2; ++tm)
            af[tm] = load_fragA(&lA[(wm * 32 + tm * 16 + hrow) * 32], hf);
        #pragma unroll
        for (int tn = 0; tn < 4; ++tn)
            bf[tn] = load_fragB(&lB[(wn * 64 + tn * 16 + hrow) * 32], hf);
        #pragma unroll
        for (int tm = 0; tm < 2; ++tm)
            #pragma unroll
            for (int tn = 0; tn < 4; ++tn)
                acc[tm][tn] = __builtin_amdgcn_wmma_f32_16x16x32_bf16(
                    false, af[tm], false, bf[tn], (short)0, acc[tm][tn], false, false);
    }

    // ---- epilogue: C/D layout — VGPR r: m = r + 8*half, n = lane%16 ----
    #pragma unroll
    for (int tm = 0; tm < 2; ++tm) {
        #pragma unroll
        for (int tn = 0; tn < 4; ++tn) {
            int n = bn + wn * 64 + tn * 16 + hrow;
            float bi = bias ? bias[n] : 0.f;
            #pragma unroll
            for (int r = 0; r < 8; ++r) {
                int m = bm + wm * 32 + tm * 16 + hf * 8 + r;
                float val = acc[tm][tn][r] + bi;
                if (ACT == 1) val = 0.5f * val * (1.f + erff(val * 0.70710678118654752f));
                if (RES) val += res[(size_t)m * N + n];
                C[(size_t)m * N + n] = val;
            }
        }
    }
}

// ---------------- driver ----------------
extern "C" void kernel_launch(void* const* d_in, const int* in_sizes, int n_in,
                              void* d_out, int out_size, void* d_ws, size_t ws_size,
                              hipStream_t stream) {
    (void)in_sizes; (void)n_in; (void)out_size; (void)ws_size;
    const int*   ids = (const int*)  d_in[0];
    const float* emb = (const float*)d_in[1];
    const float* pos = (const float*)d_in[2];
    const float* n1  = (const float*)d_in[3];
    const float* n2  = (const float*)d_in[4];
    const float* wq  = (const float*)d_in[5];  const float* bq = (const float*)d_in[6];
    const float* wk  = (const float*)d_in[7];  const float* bk = (const float*)d_in[8];
    const float* wv  = (const float*)d_in[9];  const float* bv = (const float*)d_in[10];
    const float* wo  = (const float*)d_in[11]; const float* bo = (const float*)d_in[12];
    const float* w1  = (const float*)d_in[13]; const float* b1 = (const float*)d_in[14];
    const float* w2  = (const float*)d_in[15]; const float* b2 = (const float*)d_in[16];
    const float* nf  = (const float*)d_in[17];
    float* out = (float*)d_out;

    // workspace carve-out (~80 MB)
    float* ws = (float*)d_ws;
    const size_t XD = (size_t)MTOT * DMODEL;
    float* x  = ws;
    float* h  = x  + XD;
    float* q  = h  + XD;
    float* kk = q  + XD;
    float* v  = kk + XD;
    float* o  = v  + XD;
    float* ff = o  + XD;   // MTOT x DFF

    k_embed<<<MTOT, 256, 0, stream>>>(ids, emb, pos, x);

    for (int l = 0; l < NLAYER; ++l) {
        k_rms<<<MTOT, 256, 0, stream>>>(x, n1 + (size_t)l * DMODEL, h);

        dim3 gP(DMODEL / 128, MTOT / 128);
        k_gemm<0, 0, false><<<gP, 256, 0, stream>>>(h, wq + (size_t)l * DMODEL * DMODEL,
                                                    bq + (size_t)l * DMODEL, nullptr, q,
                                                    MTOT, DMODEL, DMODEL);
        k_gemm<0, 0, false><<<gP, 256, 0, stream>>>(h, wk + (size_t)l * DMODEL * DMODEL,
                                                    bk + (size_t)l * DMODEL, nullptr, kk,
                                                    MTOT, DMODEL, DMODEL);
        k_gemm<0, 0, false><<<gP, 256, 0, stream>>>(h, wv + (size_t)l * DMODEL * DMODEL,
                                                    bv + (size_t)l * DMODEL, nullptr, v,
                                                    MTOT, DMODEL, DMODEL);

        k_attn<<<BATCH * NHEAD * SMAXC, 256, 0, stream>>>(q, kk, v, o);

        // x = x + o @ wo + bo
        k_gemm<0, 0, true><<<gP, 256, 0, stream>>>(o, wo + (size_t)l * DMODEL * DMODEL,
                                                   bo + (size_t)l * DMODEL, x, x,
                                                   MTOT, DMODEL, DMODEL);

        k_rms<<<MTOT, 256, 0, stream>>>(x, n2 + (size_t)l * DMODEL, h);

        dim3 gF1(DFFC / 128, MTOT / 128);
        k_gemm<0, 1, false><<<gF1, 256, 0, stream>>>(h, w1 + (size_t)l * DMODEL * DFFC,
                                                     b1 + (size_t)l * DFFC, nullptr, ff,
                                                     MTOT, DFFC, DMODEL);
        dim3 gF2(DMODEL / 128, MTOT / 128);
        k_gemm<0, 0, true><<<gF2, 256, 0, stream>>>(ff, w2 + (size_t)l * DFFC * DMODEL,
                                                    b2 + (size_t)l * DMODEL, x, x,
                                                    MTOT, DMODEL, DFFC);
    }

    k_rms<<<MTOT, 256, 0, stream>>>(x, nf, h);

    // logits = h @ emb^T  (emb stored [V, D] => WLAY=1)
    dim3 gL(VOCAB / 128, MTOT / 128);
    k_gemm<1, 0, false><<<gL, 256, 0, stream>>>(h, emb, nullptr, nullptr, out,
                                                MTOT, VOCAB, DMODEL);
}
